// Attn_63367947485870
// MI455X (gfx1250) — compile-verified, matrix operand order checked by reading, not run on hardware
//
#include <hip/hip_runtime.h>

// ---------------- types ----------------
typedef __attribute__((ext_vector_type(16))) __bf16 v16bf;
typedef __attribute__((ext_vector_type(8)))  __bf16 v8bf;
typedef __attribute__((ext_vector_type(8)))  float  v8f;

#define Bn   2
#define Sn   2048
#define DMn  1024
#define Hn   16
#define HDn  64
#define TPFn 128
#define NFn  16

union ABu { v16bf v; v8bf h[2]; };

// Load one 16-element bf16 operand chunk (K=32 slice) in the CDNA5 WMMA
// A/B per-lane layout: lanes 0-15 hold K {0..7, 16..23}, lanes 16-31 hold
// K {8..15, 24..31}. rowPtr points at element K=0 of this lane's row/column.
__device__ __forceinline__ v16bf load_ab(const __bf16* rowPtr, int half) {
  ABu u;
  u.h[0] = *reinterpret_cast<const v8bf*>(rowPtr + half * 8);
  u.h[1] = *reinterpret_cast<const v8bf*>(rowPtr + 16 + half * 8);
  return u.v;
}

__device__ __forceinline__ v8f wmma_bf16(v16bf a, v16bf b, v8f c) {
  return __builtin_amdgcn_wmma_f32_16x16x32_bf16(
      /*neg_a=*/false, a, /*neg_b=*/false, b,
      /*c_mod=*/(short)0, c, /*reuse_a=*/false, /*reuse_b=*/false);
}

// ---------------- fp32 -> bf16 convert ----------------
__global__ void __launch_bounds__(256)
f32_to_bf16_kernel(const float* __restrict__ in, __bf16* __restrict__ out, int n) {
  int i = blockIdx.x * 256 + threadIdx.x;
  if (i < n) out[i] = (__bf16)in[i];
}

// ---------------- shared WMMA GEMM core (ping-pong pipelined) -------------
// Computes a 16(M) x 64(N) f32 tile: C = A[M,K] * W[N,K]^T for K = DMn.
// k-loop unrolled by 2 with two explicit operand register sets so there are
// no loop-carried register copies: set0 is consumed while set1's loads are in
// flight and vice versa.
__device__ __forceinline__ void gemm_core(const __bf16* __restrict__ arow,
                                          const __bf16* const* wrow,
                                          int half, v8f c[4]) {
  v16bf a0 = load_ab(arow, half);
  v16bf b0[4];
#pragma unroll
  for (int t = 0; t < 4; ++t) b0[t] = load_ab(wrow[t], half);

  for (int kk = 0; kk < DMn; kk += 64) {
    __builtin_prefetch(arow + kk + 512, 0, 1);     // stream-ahead on A row
    // prefetch chunk kk+32 into set1 (always in range: kk <= DMn-64)
    v16bf a1 = load_ab(arow + kk + 32, half);
    v16bf b1[4];
#pragma unroll
    for (int t = 0; t < 4; ++t) b1[t] = load_ab(wrow[t] + kk + 32, half);

    // consume set0 (chunk kk)
#pragma unroll
    for (int t = 0; t < 4; ++t) c[t] = wmma_bf16(a0, b0[t], c[t]);

    // prefetch chunk kk+64 into set0 (last iter: harmless dummy reload of 0)
    const int kn = (kk + 64 < DMn) ? (kk + 64) : 0;
    a0 = load_ab(arow + kn, half);
#pragma unroll
    for (int t = 0; t < 4; ++t) b0[t] = load_ab(wrow[t] + kn, half);

    // consume set1 (chunk kk+32)
#pragma unroll
    for (int t = 0; t < 4; ++t) c[t] = wmma_bf16(a1, b1[t], c[t]);
  }
}

// ---------------- QKV projection GEMM ----------------
// C[4096 x 3072] = x_bf16[4096 x 1024] * Wqkv_bf16[3072 x 1024]^T + bqkv
// Epilogue scatters: q,k -> f32 [B,H,S,HD]; v -> bf16 transposed [B,H,HD,S].
__global__ void __launch_bounds__(128)
gemm_qkv_kernel(const __bf16* __restrict__ A, const __bf16* __restrict__ W,
                const float* __restrict__ bias,
                float* __restrict__ qf, float* __restrict__ kf,
                __bf16* __restrict__ vt) {
  const int lane = threadIdx.x & 31;
  const int wave = threadIdx.x >> 5;
  const int half = lane >> 4;
  const int l16  = lane & 15;
  const int m0 = blockIdx.y * 64 + wave * 16;   // row tile
  const int n0 = blockIdx.x * 64;               // col block (4 tiles)

  const __bf16* arow = A + (size_t)(m0 + l16) * DMn;
  const __bf16* wrow[4];
#pragma unroll
  for (int t = 0; t < 4; ++t)
    wrow[t] = W + (size_t)(n0 + t * 16 + l16) * DMn;

  v8f c[4] = {v8f{}, v8f{}, v8f{}, v8f{}};
  gemm_core(arow, wrow, half, c);

#pragma unroll
  for (int t = 0; t < 4; ++t) {
    const int ng = n0 + t * 16 + l16;   // global output column
    const float bi = bias[ng];
    const int three = ng >> 10;
    const int h  = (ng >> 6) & (Hn - 1);
    const int hd = ng & (HDn - 1);
#pragma unroll
    for (int r = 0; r < 8; ++r) {
      const int mrow = m0 + half * 8 + r;
      const int b = mrow >> 11;        // S = 2048
      const int s = mrow & (Sn - 1);
      const float val = c[t][r] + bi;
      if (three == 0)
        qf[((size_t)((b * Hn + h) * Sn + s)) * HDn + hd] = val;
      else if (three == 1)
        kf[((size_t)((b * Hn + h) * Sn + s)) * HDn + hd] = val;
      else
        vt[((size_t)((b * Hn + h) * HDn + hd)) * Sn + s] = (__bf16)val;
    }
  }
}

// ---------------- fused RMSNorm + RoPE (+q pre-scale), f32 -> bf16 ----------
// One wave per (b,h,s) row of 64; lane handles elements (lane, lane+32).
__global__ void __launch_bounds__(256)
normrope_kernel(const float* __restrict__ qf, const float* __restrict__ kf,
                const float* __restrict__ qs, const float* __restrict__ ks,
                __bf16* __restrict__ qb, __bf16* __restrict__ kb) {
  const int lane = threadIdx.x & 31;
  const int wave = threadIdx.x >> 5;
  const long ROWS = (long)Bn * Hn * Sn;         // 65536
  long g = (long)blockIdx.x * 8 + wave;         // 0 .. 2*ROWS-1
  const int isK = g >= ROWS;
  const long row = isK ? (g - ROWS) : g;
  const float* src = isK ? kf : qf;
  const float* sc  = isK ? ks : qs;
  __bf16* dst = isK ? kb : qb;
  const float outmul = isK ? 1.0f : 0.125f;     // fold HD^-0.5 into q

  const int s = (int)(row & (Sn - 1));
  const float* p = src + row * HDn;
  float e0 = p[lane], e1 = p[lane + 32];

  float ssum = e0 * e0 + e1 * e1;
  ssum += __shfl_xor(ssum, 16);
  ssum += __shfl_xor(ssum, 8);
  ssum += __shfl_xor(ssum, 4);
  ssum += __shfl_xor(ssum, 2);
  ssum += __shfl_xor(ssum, 1);
  const float rinv = rsqrtf(ssum * (1.0f / HDn) + 1e-6f);

  const float n0 = e0 * rinv * sc[lane];
  const float n1 = e1 * rinv * sc[lane + 32];
  // inv_freq = theta^(-lane/32), theta = 10000
  const float ang = (float)s * __expf(-(float)lane * (1.0f / 32.0f) * 9.210340371976184f);
  const float ca = cosf(ang), sa = sinf(ang);
  dst[row * HDn + lane]      = (__bf16)((n0 * ca - n1 * sa) * outmul);
  dst[row * HDn + lane + 32] = (__bf16)((n1 * ca + n0 * sa) * outmul);
}

// ---------------- flash attention ----------------
// 4 waves per block; waves cover 4 consecutive 16-row query tiles inside ONE
// frame (8 tiles/frame, 4 | 8) so all waves share loop bounds (legal
// __syncthreads) and stream the identical K/V range (L1 reuse).
__global__ void __launch_bounds__(128)
attn_kernel(const __bf16* __restrict__ qb, const __bf16* __restrict__ kb,
            const __bf16* __restrict__ vt, __bf16* __restrict__ attnA) {
  __shared__ __align__(16) __bf16 Pl[4][16][40];   // per-wave padded P staging

  const int lane = threadIdx.x & 31;
  const int wave = threadIdx.x >> 5;
  const int half = lane >> 4;
  const int l16  = lane & 15;
  const int gid = blockIdx.x * 4 + wave;
  const int qt = gid & 127;             // S/16
  const int h  = (gid >> 7) & (Hn - 1);
  const int b  = gid >> 11;
  const int q0 = qt * 16;
  const int qframe = q0 / TPFn;
  const int kstart = (qframe == NFn - 1) ? TPFn : 0;  // quirk: last frame masks frame 0
  const int kend   = (qframe + 1) * TPFn;             // block-causal

  const __bf16* qbase = qb + ((size_t)(b * Hn + h) * Sn) * HDn;
  const __bf16* kbase = kb + ((size_t)(b * Hn + h) * Sn) * HDn;
  const __bf16* vbase = vt + ((size_t)(b * Hn + h) * HDn) * Sn;

  const __bf16* qrow = qbase + (size_t)(q0 + l16) * HDn;
  const v16bf a0 = load_ab(qrow, half);        // hd 0..31
  const v16bf a1 = load_ab(qrow + 32, half);   // hd 32..63

  v8f o[4] = {v8f{}, v8f{}, v8f{}, v8f{}};
  float m[8], l[8];
#pragma unroll
  for (int r = 0; r < 8; ++r) { m[r] = -1e30f; l[r] = 0.0f; }

  for (int kc = kstart; kc < kend; kc += 32) {
    // ---- issue V loads first: independent, overlap score math
    v16bf vb[4];
#pragma unroll
    for (int t = 0; t < 4; ++t)
      vb[t] = load_ab(vbase + (size_t)(t * 16 + l16) * Sn + kc, half);

    // ---- K operands (columns of K^T are contiguous K rows)
    const __bf16* krow0 = kbase + (size_t)(kc + l16) * HDn;
    const __bf16* krow1 = kbase + (size_t)(kc + 16 + l16) * HDn;
    const v16bf kb00 = load_ab(krow0, half);
    const v16bf kb01 = load_ab(krow0 + 32, half);
    const v16bf kb10 = load_ab(krow1, half);
    const v16bf kb11 = load_ab(krow1 + 32, half);

    // ---- scores: S[16q x 32k] = Q * K^T
    v8f s0 = {}, s1 = {};
    s0 = wmma_bf16(a0, kb00, s0);
    s0 = wmma_bf16(a1, kb01, s0);
    s1 = wmma_bf16(a0, kb10, s1);
    s1 = wmma_bf16(a1, kb11, s1);

    // ---- online softmax (row m = half*8 + r lives on 16 lanes of same half)
#pragma unroll
    for (int r = 0; r < 8; ++r) {
      float v = fmaxf(s0[r], s1[r]);
      v = fmaxf(v, __shfl_xor(v, 1));
      v = fmaxf(v, __shfl_xor(v, 2));
      v = fmaxf(v, __shfl_xor(v, 4));
      v = fmaxf(v, __shfl_xor(v, 8));
      const float mn = fmaxf(m[r], v);
      const float scl = __expf(m[r] - mn);
      const float p0 = __expf(s0[r] - mn);
      const float p1 = __expf(s1[r] - mn);
      float rs = p0 + p1;
      rs += __shfl_xor(rs, 1);
      rs += __shfl_xor(rs, 2);
      rs += __shfl_xor(rs, 4);
      rs += __shfl_xor(rs, 8);
      l[r] = l[r] * scl + rs;
      m[r] = mn;
      s0[r] = p0; s1[r] = p1;
      o[0][r] *= scl; o[1][r] *= scl; o[2][r] *= scl; o[3][r] *= scl;
    }

    // ---- transpose P (C-layout -> A-layout) through LDS as bf16
#pragma unroll
    for (int r = 0; r < 8; ++r) {
      const int mr = half * 8 + r;
      Pl[wave][mr][l16]      = (__bf16)s0[r];
      Pl[wave][mr][l16 + 16] = (__bf16)s1[r];
    }
    __syncthreads();
    ABu pa;
    pa.h[0] = *reinterpret_cast<const v8bf*>(&Pl[wave][l16][half * 8]);
    pa.h[1] = *reinterpret_cast<const v8bf*>(&Pl[wave][l16][16 + half * 8]);
    __syncthreads();

    // ---- O += P * V   (V stored transposed: columns are contiguous)
#pragma unroll
    for (int t = 0; t < 4; ++t)
      o[t] = wmma_bf16(pa.v, vb[t], o[t]);
  }

  // ---- normalize and store as bf16 rows of attn output [B*S, DM]
  float inv[8];
#pragma unroll
  for (int r = 0; r < 8; ++r) inv[r] = 1.0f / l[r];
#pragma unroll
  for (int t = 0; t < 4; ++t) {
    const int col = h * HDn + t * 16 + l16;
#pragma unroll
    for (int r = 0; r < 8; ++r) {
      const int mrow = q0 + half * 8 + r;
      attnA[((size_t)(b * Sn + mrow)) * DMn + col] = (__bf16)(o[t][r] * inv[r]);
    }
  }
}

// ---------------- output projection GEMM ----------------
// out[4096 x 1024] = attnA_bf16[4096 x 1024] * Wout_bf16[1024 x 1024]^T + bout
__global__ void __launch_bounds__(128)
gemm_out_kernel(const __bf16* __restrict__ A, const __bf16* __restrict__ W,
                const float* __restrict__ bias, float* __restrict__ out) {
  const int lane = threadIdx.x & 31;
  const int wave = threadIdx.x >> 5;
  const int half = lane >> 4;
  const int l16  = lane & 15;
  const int m0 = blockIdx.y * 64 + wave * 16;
  const int n0 = blockIdx.x * 64;

  const __bf16* arow = A + (size_t)(m0 + l16) * DMn;
  const __bf16* wrow[4];
#pragma unroll
  for (int t = 0; t < 4; ++t)
    wrow[t] = W + (size_t)(n0 + t * 16 + l16) * DMn;

  v8f c[4] = {v8f{}, v8f{}, v8f{}, v8f{}};
  gemm_core(arow, wrow, half, c);

#pragma unroll
  for (int t = 0; t < 4; ++t) {
    const int ng = n0 + t * 16 + l16;
    const float bi = bias[ng];
#pragma unroll
    for (int r = 0; r < 8; ++r) {
      const int mrow = m0 + half * 8 + r;
      out[(size_t)mrow * DMn + ng] = c[t][r] + bi;
    }
  }
}

// ---------------- launcher ----------------
extern "C" void kernel_launch(void* const* d_in, const int* in_sizes, int n_in,
                              void* d_out, int out_size, void* d_ws, size_t ws_size,
                              hipStream_t stream) {
  const float* x     = (const float*)d_in[0];  // [B,S,DM]
  const float* Wqkv  = (const float*)d_in[1];  // [3DM,DM]
  const float* bqkv  = (const float*)d_in[2];  // [3DM]
  const float* qsc   = (const float*)d_in[3];  // [HD]
  const float* ksc   = (const float*)d_in[4];  // [HD]
  const float* Wout  = (const float*)d_in[5];  // [DM,DM]
  const float* bout  = (const float*)d_in[6];  // [DM]
  float* out = (float*)d_out;

  const size_t NTOK = (size_t)Bn * Sn;          // 4096
  const size_t NQKV = NTOK * DMn;               // 4M  (= B*H*S*HD)

  char* w = (char*)d_ws;
  __bf16* xb  = (__bf16*)w; w += NQKV * 2;                  // x bf16
  __bf16* wqb = (__bf16*)w; w += (size_t)3 * DMn * DMn * 2; // Wqkv bf16
  __bf16* wob = (__bf16*)w; w += (size_t)DMn * DMn * 2;     // Wout bf16
  float*  qf  = (float*)w;  w += NQKV * 4;                  // q f32 [B,H,S,HD]
  float*  kf  = (float*)w;  w += NQKV * 4;                  // k f32 [B,H,S,HD]
  __bf16* vtb = (__bf16*)w; w += NQKV * 2;                  // v bf16 [B,H,HD,S]
  __bf16* qbb = (__bf16*)w; w += NQKV * 2;                  // q bf16 (normed/roped/scaled)
  __bf16* kbb = (__bf16*)w; w += NQKV * 2;                  // k bf16
  __bf16* atA = (__bf16*)w; w += NQKV * 2;                  // attn out bf16 [B*S, DM]

  // 1) fp32 -> bf16 conversions
  f32_to_bf16_kernel<<<(int)((NQKV + 255) / 256), 256, 0, stream>>>(x, xb, (int)NQKV);
  f32_to_bf16_kernel<<<(3 * DMn * DMn + 255) / 256, 256, 0, stream>>>(Wqkv, wqb, 3 * DMn * DMn);
  f32_to_bf16_kernel<<<(DMn * DMn + 255) / 256, 256, 0, stream>>>(Wout, wob, DMn * DMn);

  // 2) QKV projection (WMMA bf16), epilogue splits q/k (f32) and v (bf16, transposed)
  gemm_qkv_kernel<<<dim3(3 * DMn / 64, NTOK / 64), 128, 0, stream>>>(xb, wqb, bqkv, qf, kf, vtb);

  // 3) fused RMSNorm + RoPE (+ 1/sqrt(HD) folded into q), write bf16
  normrope_kernel<<<(2 * Bn * Hn * Sn) / 8, 256, 0, stream>>>(qf, kf, qsc, ksc, qbb, kbb);

  // 4) flash attention with frame-block-causal bounds (quirk via kstart)
  attn_kernel<<<(Bn * Hn * (Sn / 16)) / 4, 128, 0, stream>>>(qbb, kbb, vtb, atA);

  // 5) output projection (WMMA bf16) -> fp32 result
  gemm_out_kernel<<<dim3(DMn / 64, NTOK / 64), 128, 0, stream>>>(atA, wob, bout, out);
}
